// SelfAttention_60095182406127
// MI455X (gfx1250) — compile-verified, matrix operand order checked by reading, not run on hardware
//
#include <hip/hip_runtime.h>

typedef __bf16 bf16;
typedef __attribute__((ext_vector_type(16))) __bf16 v16bf;
typedef __attribute__((ext_vector_type(8)))  float  v8f;

#define HEADS 16
#define DH    64
#define EMB   1024
#define SEQ   4096
#define BATCH 2

// ---- fragment loaders (CDNA5 16-bit A/B layout, ISA 7.12.2) ----
// A (16x32, MxK): lane m = lane&15; per-lane 16 bf16 = K chunks
// [half*8 .. half*8+7] and [half*8+16 .. half*8+23].  B (32x16, KxN) is the
// same pattern with lane -> column N.  Both load as two contiguous 16B chunks.

static __device__ inline v16bf load_frag_f32(const float* __restrict__ base, int ld, int lane) {
    const int m = lane & 15, half = lane >> 4;
    const float* p0 = base + (size_t)m * ld + half * 8;
    v16bf a;
#pragma unroll
    for (int i = 0; i < 8; ++i) {
        a[i]     = (__bf16)p0[i];
        a[i + 8] = (__bf16)p0[i + 16];
    }
    return a;
}

static __device__ inline v16bf load_frag_bf16(const bf16* __restrict__ base, int ld, int lane) {
    const int m = lane & 15, half = lane >> 4;
    const bf16* p0 = base + (size_t)m * ld + half * 8;
    v16bf a;
#pragma unroll
    for (int i = 0; i < 8; ++i) {
        a[i]     = p0[i];
        a[i + 8] = p0[i + 16];
    }
    return a;
}

// CDNA5 async global->LDS copy (ASYNCcnt-tracked, bypasses VGPRs).
// Generic pointers to __shared__ carry the LDS byte offset in their low 32
// bits (aperture truncation, ISA 10.2).
static __device__ inline void async_ld128(const bf16* lds_dst, const bf16* g_src) {
    const unsigned int       l = (unsigned int)(size_t)lds_dst;
    const unsigned long long g = (unsigned long long)(size_t)g_src;
    asm volatile("global_load_async_to_lds_b128 %0, %1, off"
                 :: "v"(l), "v"(g) : "memory");
}
static __device__ inline void wait_async0() {
    asm volatile("s_wait_asynccnt 0x0" ::: "memory");
}

static __device__ inline v8f wmma_bf16(v16bf a, v16bf b, v8f c) {
    return __builtin_amdgcn_wmma_f32_16x16x32_bf16(false, a, false, b, (short)0, c, false, false);
}

// ---------------------------------------------------------------------------
// Kernel 1: per-head linear projection  out[m][o] = scale * sum_d X[m][d]*W[o][d]
// X viewed as [N*L*H, 64] rows; W is [64,64]; out bf16.
// ---------------------------------------------------------------------------
__global__ __launch_bounds__(256) void proj_bf16_kernel(
    const float* __restrict__ X, const float* __restrict__ W,
    bf16* __restrict__ out, float scale) {
    const int lane = threadIdx.x & 31;
    const int wv   = threadIdx.x >> 5;
    const int rowTile = blockIdx.x * 2 + (wv >> 2);
    const int colTile = wv & 3;
    const int m0 = rowTile * 16;

    v8f c = {};
#pragma unroll
    for (int ks = 0; ks < 2; ++ks) {
        v16bf a = load_frag_f32(X + (size_t)m0 * DH + ks * 32, DH, lane);
        v16bf b = load_frag_f32(W + (size_t)(colTile * 16) * DH + ks * 32, DH, lane);
        c = wmma_bf16(a, b, c);
    }
    const int half = lane >> 4, nn = lane & 15;
#pragma unroll
    for (int r = 0; r < 8; ++r)
        out[(size_t)(m0 + r + 8 * half) * DH + colTile * 16 + nn] = (bf16)(c[r] * scale);
}

// ---------------------------------------------------------------------------
// Kernel 1b: transpose projected V per (n,h): [l][d] -> [d][l], LDS-tiled,
// both sides 16B-coalesced.  Makes attention's V staging pure async DMA.
// ---------------------------------------------------------------------------
__global__ __launch_bounds__(256) void transpose_v_kernel(
    const bf16* __restrict__ vp, bf16* __restrict__ vt) {
    __shared__ bf16 tile[64][72];   // +8 pad vs bank conflicts
    const int n = blockIdx.z, h = blockIdx.y;
    const int l0 = blockIdx.x * 64;
    const int tid = threadIdx.x;

    const int lrow = tid >> 2;               // 0..63
#pragma unroll
    for (int it = 0; it < 2; ++it) {
        const int doff = ((tid & 3) * 2 + it) * 8;
        union { uint4 u; bf16 e[8]; } d;
        d.u = *(const uint4*)(vp + ((size_t)((n * SEQ + l0 + lrow) * HEADS + h)) * DH + doff);
#pragma unroll
        for (int i = 0; i < 8; ++i) tile[lrow][doff + i] = d.e[i];
    }
    __syncthreads();
    const int drow = tid >> 2;               // 0..63
#pragma unroll
    for (int it = 0; it < 2; ++it) {
        const int loff = ((tid & 3) * 2 + it) * 8;
        union { uint4 u; bf16 e[8]; } d;
#pragma unroll
        for (int i = 0; i < 8; ++i) d.e[i] = tile[loff + i][drow];
        *(uint4*)(vt + ((size_t)(n * HEADS + h) * DH + drow) * SEQ + l0 + loff) = d.u;
    }
}

// ---------------------------------------------------------------------------
// Kernel 2: flash attention per (n, h).  Block = 8 waves x 32 query rows
// (two 16-row tiles per wave -> K/V fragments reused twice).
// Double-buffered 32-key LDS tiles staged entirely by async global->LDS DMA.
// ---------------------------------------------------------------------------
__global__ __launch_bounds__(256) void attn_kernel(
    const bf16* __restrict__ qp, const bf16* __restrict__ kp,
    const bf16* __restrict__ vt, bf16* __restrict__ op) {
    __shared__ bf16 sK [2][32 * DH];    // [buf][key][d]   (2 x 4 KB)
    __shared__ bf16 sVt[2][DH * 32];    // [buf][d][key]   (2 x 4 KB)
    __shared__ bf16 sP [8][16 * 32];    // per-wave P tile (8 KB)

    const int tid  = threadIdx.x;
    const int lane = tid & 31;
    const int wv   = tid >> 5;
    const int half = lane >> 4, nn = lane & 15;

    const int n = blockIdx.z;
    const int h = blockIdx.y;
    const int qbase = blockIdx.x * 256 + wv * 32;

    // Q fragments for two 16-row tiles (1/sqrt(E) folded into Q projection)
    v16bf qa[2][2];
#pragma unroll
    for (int t = 0; t < 2; ++t) {
        const bf16* qrow = qp + ((size_t)((n * SEQ + qbase + t * 16) * HEADS + h)) * DH;
        qa[t][0] = load_frag_bf16(qrow,      HEADS * DH, lane);
        qa[t][1] = load_frag_bf16(qrow + 32, HEADS * DH, lane);
    }

    float mrow[2][8], lrow[2][8];
    v8f o[2][4] = {};
#pragma unroll
    for (int t = 0; t < 2; ++t)
#pragma unroll
        for (int r = 0; r < 8; ++r) { mrow[t][r] = -1e30f; lrow[t][r] = 0.f; }

    // async staging: K rows ([key][d], 128B/key) + V^T rows ([d][l], 64B/tile-row)
    auto stage_tile = [&](int buf, int kb) {
        const int keyId = tid >> 3, keyChunk = tid & 7;
        const size_t ka =
            ((size_t)(n * SEQ + kb + keyId) * HEADS + h) * DH + keyChunk * 8;
        async_ld128(&sK[buf][keyId * DH + keyChunk * 8], kp + ka);
        const int dRow = tid >> 2, part = tid & 3;
        const size_t va =
            ((size_t)(n * HEADS + h) * DH + dRow) * SEQ + kb + part * 8;
        async_ld128(&sVt[buf][dRow * 32 + part * 8], vt + va);
    };

    stage_tile(0, 0);  // prologue

    for (int kb = 0; kb < SEQ; kb += 32) {
        const int cur = (kb >> 5) & 1;
        wait_async0();        // own async writes to tile `cur` are in LDS
        __syncthreads();      // all tile writes visible; prev tile consumed
        if (kb + 32 < SEQ) stage_tile(cur ^ 1, kb + 32);

        // K fragments, shared by both query tiles
        v16bf kf0[2], kf1[2];
#pragma unroll
        for (int jt = 0; jt < 2; ++jt) {
            kf0[jt] = load_frag_bf16(&sK[cur][jt * 16 * DH],      DH, lane);
            kf1[jt] = load_frag_bf16(&sK[cur][jt * 16 * DH + 32], DH, lane);
        }

#pragma unroll
        for (int t = 0; t < 2; ++t) {
            // S = Q K^T (two 16-key subtiles, K=64 contraction)
            v8f s[2];
#pragma unroll
            for (int jt = 0; jt < 2; ++jt) {
                v8f acc = {};
                acc = wmma_bf16(qa[t][0], kf0[jt], acc);
                acc = wmma_bf16(qa[t][1], kf1[jt], acc);
                s[jt] = acc;
            }

            // online softmax: lane holds 8 rows x 1 col per subtile;
            // rows span a 16-lane half -> shfl_xor masks 1,2,4,8.
            float alpha[8];
#pragma unroll
            for (int r = 0; r < 8; ++r) {
                float mnew = fmaxf(s[0][r], s[1][r]);
#pragma unroll
                for (int msk = 1; msk < 16; msk <<= 1)
                    mnew = fmaxf(mnew, __shfl_xor(mnew, msk, 32));
                const float mi = fmaxf(mrow[t][r], mnew);
                const float a  = __expf(mrow[t][r] - mi);
                mrow[t][r] = mi;
                const float p0 = __expf(s[0][r] - mi);
                const float p1 = __expf(s[1][r] - mi);
                s[0][r] = p0; s[1][r] = p1;
                float rs = p0 + p1;
#pragma unroll
                for (int msk = 1; msk < 16; msk <<= 1)
                    rs += __shfl_xor(rs, msk, 32);
                lrow[t][r] = lrow[t][r] * a + rs;
                alpha[r] = a;
            }
#pragma unroll
            for (int dt = 0; dt < 4; ++dt)
#pragma unroll
                for (int r = 0; r < 8; ++r) o[t][dt][r] *= alpha[r];

            // spill P (C-fragment layout) to per-wave LDS, reload as A-frag
#pragma unroll
            for (int r = 0; r < 8; ++r) {
                sP[wv][(r + 8 * half) * 32 + nn]      = (bf16)s[0][r];
                sP[wv][(r + 8 * half) * 32 + 16 + nn] = (bf16)s[1][r];
            }
            const v16bf pa = load_frag_bf16(&sP[wv][0], 32, lane);

            // O += P V (V^T in LDS gives contiguous B-fragment chunks)
#pragma unroll
            for (int dt = 0; dt < 4; ++dt)
                o[t][dt] = wmma_bf16(pa, load_frag_bf16(&sVt[cur][dt * 16 * 32], 32, lane), o[t][dt]);
        }
    }

    // normalize and store (bf16, [N*L, E] with head offset)
#pragma unroll
    for (int t = 0; t < 2; ++t) {
        bf16* obase = op + ((size_t)((n * SEQ + qbase + t * 16) * HEADS + h)) * DH;
#pragma unroll
        for (int r = 0; r < 8; ++r) {
            const float inv = 1.f / lrow[t][r];
            const size_t row = (size_t)(r + 8 * half) * (HEADS * DH);
#pragma unroll
            for (int dt = 0; dt < 4; ++dt)
                obase[row + dt * 16 + nn] = (bf16)(o[t][dt][r] * inv);
        }
    }
}

// ---------------------------------------------------------------------------
// Kernel 3a: one-shot fp32 -> bf16 conversion of Wfc (stays hot in L2)
// ---------------------------------------------------------------------------
__global__ __launch_bounds__(256) void cvt_f32_bf16_kernel(
    const float* __restrict__ in, bf16* __restrict__ out) {
    const size_t i = ((size_t)blockIdx.x * 256 + threadIdx.x) * 8;
    union { uint4 u; bf16 e[8]; } o;
#pragma unroll
    for (int j = 0; j < 8; ++j) o.e[j] = (bf16)in[i + j];
    *(uint4*)(out + i) = o.u;
}

// ---------------------------------------------------------------------------
// Kernel 3b: final FC  out = A @ Wfc^T + bfc   (M=8192, K=N=1024)
// Each wave computes a 16x64 strip: A-fragment reused by 4 WMMAs per k-step.
// ---------------------------------------------------------------------------
__global__ __launch_bounds__(256) void fc_kernel(
    const bf16* __restrict__ A, const bf16* __restrict__ Wb,
    const float* __restrict__ bias, float* __restrict__ out) {
    const int lane = threadIdx.x & 31;
    const int wv   = threadIdx.x >> 5;
    const int tile = blockIdx.x * 8 + wv;
    const int rowTile = tile >> 4;          // EMB/64 = 16 column groups
    const int colGrp  = tile & 15;
    const int m0 = rowTile * 16, n0 = colGrp * 64;

    v8f c[4] = {{}, {}, {}, {}};
#pragma unroll 2
    for (int k0 = 0; k0 < EMB; k0 += 32) {
        const v16bf a = load_frag_bf16(A + (size_t)m0 * EMB + k0, EMB, lane);
#pragma unroll
        for (int j = 0; j < 4; ++j) {
            v16bf b = load_frag_bf16(Wb + (size_t)(n0 + j * 16) * EMB + k0, EMB, lane);
            c[j] = wmma_bf16(a, b, c[j]);
        }
    }
    const int half = lane >> 4, nn = lane & 15;
#pragma unroll
    for (int j = 0; j < 4; ++j) {
        const float bv = bias[n0 + j * 16 + nn];
#pragma unroll
        for (int r = 0; r < 8; ++r)
            out[(size_t)(m0 + r + 8 * half) * EMB + n0 + j * 16 + nn] = c[j][r] + bv;
    }
}

// ---------------------------------------------------------------------------
extern "C" void kernel_launch(void* const* d_in, const int* in_sizes, int n_in,
                              void* d_out, int out_size, void* d_ws, size_t ws_size,
                              hipStream_t stream) {
    const float* values  = (const float*)d_in[0];
    const float* keys    = (const float*)d_in[1];
    const float* queries = (const float*)d_in[2];
    const float* Wv      = (const float*)d_in[3];
    const float* Wk      = (const float*)d_in[4];
    const float* Wq      = (const float*)d_in[5];
    const float* Wfc     = (const float*)d_in[6];
    const float* bfc     = (const float*)d_in[7];
    float* out = (float*)d_out;

    const size_t elems = (size_t)BATCH * SEQ * EMB;  // 8.4M per buffer
    bf16* vp = (bf16*)d_ws;
    bf16* kp = vp + elems;
    bf16* qp = kp + elems;
    bf16* ap = qp + elems;
    bf16* vt = ap + elems;                 // V^T [N,H,D,L] (16 MB)
    bf16* wb = vt + elems;                 // Wfc bf16 (2 MB); total ws ~82 MB

    const int Mrows = BATCH * SEQ * HEADS;  // 131072 rows of 64
    dim3 blk(256);

    proj_bf16_kernel<<<Mrows / 32, blk, 0, stream>>>(values,  Wv, vp, 1.0f);
    proj_bf16_kernel<<<Mrows / 32, blk, 0, stream>>>(keys,    Wk, kp, 1.0f);
    proj_bf16_kernel<<<Mrows / 32, blk, 0, stream>>>(queries, Wq, qp, 1.0f / 32.0f);
    cvt_f32_bf16_kernel<<<(EMB * EMB) / (256 * 8), blk, 0, stream>>>(Wfc, wb);
    transpose_v_kernel<<<dim3(SEQ / 64, HEADS, BATCH), blk, 0, stream>>>(vp, vt);

    attn_kernel<<<dim3(SEQ / 256, HEADS, BATCH), blk, 0, stream>>>(qp, kp, vt, ap);

    const int fcStrips = (BATCH * SEQ / 16) * (EMB / 64);  // 8192 wave-strips
    fc_kernel<<<fcStrips / 8, blk, 0, stream>>>(ap, wb, bfc, out);
}
